// Oracle850BRouter_50697793962043
// MI455X (gfx1250) — matrix operand order repaired and checked by reading, not compiled, not run on hardware
//
#include <hip/hip_runtime.h>
#include <hip/hip_bf16.h>

typedef __attribute__((ext_vector_type(16))) __bf16 v16bf;
typedef __attribute__((ext_vector_type(8)))  float  v8f;

#define D_MODEL     2048
#define NUM_EXPERTS 64
#define TOKENS      16384      // 8 * 2048
#define CAPACITY    2560       // int(1.25 * 2048)

// d_out flat layout (all float32, reference tuple order)
#define OUT_PROBS   0          // [16384][2]
#define OUT_IDX     32768      // [16384][2]
#define OUT_LB      65536      // scalar
#define OUT_ASSIGN  65537      // [64][2560]

// ---------------------------------------------------------------------------
// Kernel 0: zero the per-expert probability accumulators in workspace
// ---------------------------------------------------------------------------
__global__ void moe_init(float* __restrict__ gesum) {
    gesum[threadIdx.x] = 0.0f;
}

// ---------------------------------------------------------------------------
// Kernel 1: router GEMM (bf16 WMMA) + top-2 + softmax + masked scores + lb sums
// grid = 256 blocks * 128 threads (4 waves); each wave owns a 16-token tile.
// M=16384, N=64, K=2048: HBM-bound on x (134 MB ~ 5.8us at 23.3 TB/s), so
// bf16 matrix math is free; W_router (512 KB) stays L2-resident.
// ---------------------------------------------------------------------------
__launch_bounds__(128)
__global__ void moe_router_gemm_topk(const float* __restrict__ x,
                                     const float* __restrict__ W,
                                     float* __restrict__ out,
                                     float* __restrict__ scores,
                                     float* __restrict__ gesum) {
    __shared__ float slog[4][16][65];      // per-wave 16 tokens x 64 experts, padded
    __shared__ float sesum[NUM_EXPERTS];

    const int tid  = threadIdx.x;
    const int wave = tid >> 5;
    const int lane = tid & 31;
    const int half = lane >> 4;            // K-half selector (wave32 WMMA layout)
    const int l16  = lane & 15;

    if (tid < NUM_EXPERTS) sesum[tid] = 0.0f;

    const int tile = blockIdx.x * 4 + wave;      // 16-token tile index
    const int m0   = tile * 16;

    const float* xrow = x + (size_t)(m0 + l16) * D_MODEL;

    v8f acc[4] = {};   // 4 N-tiles of 16 -> experts 0..63

    #pragma unroll 2
    for (int k0 = 0; k0 < D_MODEL; k0 += 32) {
        // streaming prefetch ~1KB ahead in this row (global_prefetch_b8)
        if (k0 + 256 < D_MODEL) __builtin_prefetch(xrow + k0 + 256, 0, 3);

        // ---- A fragment: 16x32 bf16, ISA layout: lanes 0-15 rows M, K
        // interleave VGPR0-3 = K[0..7|8..15], VGPR4-7 = K[16..23|24..31]
        const int ka = k0 + half * 8;
        float4 a0 = *(const float4*)(xrow + ka);
        float4 a1 = *(const float4*)(xrow + ka + 4);
        float4 a2 = *(const float4*)(xrow + ka + 16);
        float4 a3 = *(const float4*)(xrow + ka + 20);
        v16bf A;
        A[0]=(__bf16)a0.x;  A[1]=(__bf16)a0.y;  A[2]=(__bf16)a0.z;  A[3]=(__bf16)a0.w;
        A[4]=(__bf16)a1.x;  A[5]=(__bf16)a1.y;  A[6]=(__bf16)a1.z;  A[7]=(__bf16)a1.w;
        A[8]=(__bf16)a2.x;  A[9]=(__bf16)a2.y;  A[10]=(__bf16)a2.z; A[11]=(__bf16)a2.w;
        A[12]=(__bf16)a3.x; A[13]=(__bf16)a3.y; A[14]=(__bf16)a3.z; A[15]=(__bf16)a3.w;

        // ---- B fragments: 32x16 bf16 per N-tile; column N=lane, K interleaved
        // like A: elems 0-7 = K[0..7|8..15], elems 8-15 = K[16..23|24..31]
        const int kb = k0 + half * 8;
        #pragma unroll
        for (int t = 0; t < 4; ++t) {
            const float* wrow = W + (size_t)(t * 16 + l16) * D_MODEL + kb;
            float4 b0 = *(const float4*)(wrow);
            float4 b1 = *(const float4*)(wrow + 4);
            float4 b2 = *(const float4*)(wrow + 16);
            float4 b3 = *(const float4*)(wrow + 20);
            v16bf Bv;
            Bv[0]=(__bf16)b0.x;  Bv[1]=(__bf16)b0.y;  Bv[2]=(__bf16)b0.z;  Bv[3]=(__bf16)b0.w;
            Bv[4]=(__bf16)b1.x;  Bv[5]=(__bf16)b1.y;  Bv[6]=(__bf16)b1.z;  Bv[7]=(__bf16)b1.w;
            Bv[8]=(__bf16)b2.x;  Bv[9]=(__bf16)b2.y;  Bv[10]=(__bf16)b2.z; Bv[11]=(__bf16)b2.w;
            Bv[12]=(__bf16)b3.x; Bv[13]=(__bf16)b3.y; Bv[14]=(__bf16)b3.z; Bv[15]=(__bf16)b3.w;

            acc[t] = __builtin_amdgcn_wmma_f32_16x16x32_bf16(
                false, A, false, Bv, (short)0, acc[t], false, false);
        }
    }

    // ---- scatter C (16x16 f32 layout: VGPR r -> M=r (lanes 0-15) / r+8) to LDS
    #pragma unroll
    for (int t = 0; t < 4; ++t) {
        #pragma unroll
        for (int r = 0; r < 8; ++r) {
            slog[wave][r + 8 * half][t * 16 + l16] = acc[t][r];
        }
    }
    __syncthreads();

    // ---- per-token reductions: 64 tokens per block, one thread each
    if (tid < 64) {
        const int w  = tid >> 4;
        const int tr = tid & 15;
        const int tok = blockIdx.x * 64 + tid;

        float l1 = -__builtin_inff(), l2 = -__builtin_inff();
        int   i1 = 0, i2 = 0;
        #pragma unroll 1
        for (int e = 0; e < NUM_EXPERTS; ++e) {
            float v = slog[w][tr][e];
            if (v > l1)      { l2 = l1; i2 = i1; l1 = v; i1 = e; }
            else if (v > l2) { l2 = v; i2 = e; }
        }

        float denom = 0.0f;
        #pragma unroll 1
        for (int e = 0; e < NUM_EXPERTS; ++e)
            denom += __expf(slog[w][tr][e] - l1);
        const float inv = 1.0f / denom;

        #pragma unroll 1
        for (int e = 0; e < NUM_EXPERTS; ++e) {
            float p = __expf(slog[w][tr][e] - l1) * inv;
            scores[(size_t)tok * NUM_EXPERTS + e] =
                (e == i1 || e == i2) ? p : -__builtin_inff();
            atomicAdd(&sesum[e], p);
        }

        // softmax over the two top logits
        float r2 = __expf(l2 - l1);
        float pd = 1.0f / (1.0f + r2);
        out[OUT_PROBS + tok * 2 + 0] = pd;
        out[OUT_PROBS + tok * 2 + 1] = r2 * pd;
        out[OUT_IDX   + tok * 2 + 0] = (float)i1;
        out[OUT_IDX   + tok * 2 + 1] = (float)i2;
    }
    __syncthreads();
    if (tid < NUM_EXPERTS) atomicAdd(&gesum[tid], sesum[tid]);
}

// ---------------------------------------------------------------------------
// Kernel 2: load-balance loss from per-expert prob sums
// ---------------------------------------------------------------------------
__global__ void moe_lb_loss(const float* __restrict__ gesum,
                            float* __restrict__ out) {
    __shared__ float red[NUM_EXPERTS];
    const int tid = threadIdx.x;
    float p = gesum[tid] * (1.0f / (float)TOKENS);
    red[tid] = p * __logf(p + 1e-8f);
    __syncthreads();
    for (int s = 32; s > 0; s >>= 1) {
        if (tid < s) red[tid] += red[tid + s];
        __syncthreads();
    }
    if (tid == 0) out[OUT_LB] = 0.01f * red[0];
}

// ---------------------------------------------------------------------------
// Kernel 3: capacity dispatch. One workgroup per expert; bitonic sort of
// 16384 packed (score||~token) keys in 128 KB dynamic LDS (CDNA5: 320 KB/WGP),
// descending; ties resolved toward lower token index (jax top_k semantics).
// ---------------------------------------------------------------------------
__launch_bounds__(1024)
__global__ void moe_assign(const float* __restrict__ scores,
                           float* __restrict__ out) {
    extern __shared__ unsigned long long key[];   // TOKENS entries = 128 KB
    const int e   = blockIdx.x;
    const int tid = threadIdx.x;

    for (int t = tid; t < TOKENS; t += 1024) {
        float s = scores[(size_t)t * NUM_EXPERTS + e];
        unsigned u = __float_as_uint(s);
        u = (u & 0x80000000u) ? ~u : (u | 0x80000000u);  // order-preserving map
        key[t] = ((unsigned long long)u << 32) | (unsigned)(~t);
    }
    __syncthreads();

    for (unsigned k = 2; k <= TOKENS; k <<= 1) {
        for (unsigned j = k >> 1; j > 0; j >>= 1) {
            for (unsigned i = tid; i < TOKENS; i += 1024) {
                unsigned ij = i ^ j;
                if (ij > i) {
                    unsigned long long a = key[i];
                    unsigned long long b = key[ij];
                    bool desc = ((i & k) == 0);
                    bool sw = desc ? (a < b) : (a > b);
                    if (sw) { key[i] = b; key[ij] = a; }
                }
            }
            __syncthreads();
        }
    }

    for (int s = tid; s < CAPACITY; s += 1024) {
        unsigned long long kk = key[s];
        unsigned ord = (unsigned)(kk >> 32);
        // any real prob maps to ord >= 0x80000000; -inf maps below -> pad -1
        float v = (ord >= 0x80000000u) ? (float)(~(unsigned)kk) : -1.0f;
        out[OUT_ASSIGN + e * CAPACITY + s] = v;
    }
}

// ---------------------------------------------------------------------------
extern "C" void kernel_launch(void* const* d_in, const int* in_sizes, int n_in,
                              void* d_out, int out_size, void* d_ws, size_t ws_size,
                              hipStream_t stream) {
    const float* x = (const float*)d_in[0];   // [8,2048,2048]
    const float* W = (const float*)d_in[1];   // [64,2048]
    float* out = (float*)d_out;

    float* scores = (float*)d_ws;                              // [16384][64] f32
    float* gesum  = scores + (size_t)TOKENS * NUM_EXPERTS;     // [64] f32

    moe_init<<<1, NUM_EXPERTS, 0, stream>>>(gesum);
    moe_router_gemm_topk<<<256, 128, 0, stream>>>(x, W, out, scores, gesum);
    moe_lb_loss<<<1, NUM_EXPERTS, 0, stream>>>(gesum, out);
    moe_assign<<<NUM_EXPERTS, 1024,
                 (size_t)TOKENS * sizeof(unsigned long long), stream>>>(scores, out);
}
